// GatherMatmulScatterLinear_85873576116768
// MI455X (gfx1250) — matrix operand order repaired and checked by reading, not compiled, not run on hardware
//
#include <hip/hip_runtime.h>

// ---------------- problem constants (from reference) ----------------
#define D_FF     11008
#define D_MODEL  4096
#define K_AL     5504    // alive input channels  (== 172 * 32, exact)
#define N_AL     2048    // alive output channels (== 16 * 128, exact)
#define M_ROWS   8192    // BATCH * SEQ           (== 64 * 128, exact)

typedef __bf16 bf16;
typedef __attribute__((ext_vector_type(16))) __bf16 v16bf;
typedef __attribute__((ext_vector_type(8)))  __bf16 bf16x8;  // one b128
typedef __attribute__((ext_vector_type(8)))  float  v8f;
typedef __attribute__((ext_vector_type(4)))  float  f32x4;

// ---------------- GEMM tiling ----------------
#define BM   128
#define BN   128
#define BK   32
#define LDSS 56   // LDS row stride in bf16 (32 data + 24 pad): 112B rows,
                  // 16B-aligned b128 accesses, conflict-free 16-row frag reads
#define NKSTEPS (K_AL / BK)   // 172

union Frag { v16bf v; bf16x8 h[2]; };

// ---------------- output zeroing ----------------
__global__ __launch_bounds__(256) void zero_kernel(f32x4* __restrict__ p, long n4) {
    long i = (long)blockIdx.x * blockDim.x + threadIdx.x;
    if (i < n4) {
        f32x4 z; z[0] = 0.f; z[1] = 0.f; z[2] = 0.f; z[3] = 0.f;
        p[i] = z;
    }
}

// ---------------- gather + fp32 -> bf16 hi/lo split ----------------
__global__ __launch_bounds__(256) void gather_x_kernel(
    const float* __restrict__ x, const int* __restrict__ alive_in,
    bf16* __restrict__ Ahi, bf16* __restrict__ Alo)
{
    long g = (long)blockIdx.x * blockDim.x + threadIdx.x;
    if (g >= (long)M_ROWS * K_AL) return;
    int  col = (int)(g % K_AL);
    long row = g / K_AL;
    float v  = x[row * D_FF + alive_in[col]];
    bf16 hi  = (bf16)v;
    bf16 lo  = (bf16)(v - (float)hi);
    Ahi[g] = hi;
    Alo[g] = lo;
}

__global__ __launch_bounds__(256) void gather_w_kernel(
    const float* __restrict__ W, const int* __restrict__ alive_in,
    const int* __restrict__ alive_out,
    bf16* __restrict__ Bhi, bf16* __restrict__ Blo)
{
    long g = (long)blockIdx.x * blockDim.x + threadIdx.x;
    if (g >= (long)N_AL * K_AL) return;
    int col = (int)(g % K_AL);
    int row = (int)(g / K_AL);
    float v = W[(long)alive_out[row] * D_FF + alive_in[col]];
    bf16 hi = (bf16)v;
    bf16 lo = (bf16)(v - (float)hi);
    Bhi[g] = hi;
    Blo[g] = lo;
}

// ------- bf16x3 WMMA GEMM, double-buffered LDS, bias + column scatter -------
__global__ __launch_bounds__(256) void gemm_kernel(
    const bf16* __restrict__ Ahi, const bf16* __restrict__ Alo,
    const bf16* __restrict__ Bhi, const bf16* __restrict__ Blo,
    const float* __restrict__ bias, const int* __restrict__ alive_out,
    float* __restrict__ out)
{
    __shared__ bf16 sAh[2][BM * LDSS];
    __shared__ bf16 sAl[2][BM * LDSS];
    __shared__ bf16 sBh[2][BN * LDSS];
    __shared__ bf16 sBl[2][BN * LDSS];

    const int tid  = threadIdx.x;         // 0..255, 8 wave32s
    const int lane = tid & 31;
    const int wave = tid >> 5;
    const int wm   = wave & 3;            // wave M offset: wm*32 (2 subtiles)
    const int wn   = wave >> 2;           // wave N offset: wn*64 (4 subtiles)
    const long mBlock = (long)blockIdx.x * BM;
    const int  nBlock = (int)blockIdx.y * BN;

    // global->LDS staging: thread t covers rows ldRow & ldRow+64, one 8-elem segment
    const int ldRow = tid >> 2;           // 0..63
    const int ldCol = (tid & 3) * 8;      // 0,8,16,24

    // fragment addressing (per ISA VGPR layouts, wave32)
    const int half = lane >> 4;           // lane half selects K split
    const int lr   = lane & 15;           // M (for A) / N (for B,C)
    const int ka0  = half * 8;            // A: K = {0..7,16..23} | {8..15,24..31}
    const int kb0  = half * 8;            // B: K = {0..7,16..23} | {8..15,24..31}

    v8f acc[2][4];
#pragma unroll
    for (int m = 0; m < 2; ++m)
#pragma unroll
        for (int n = 0; n < 4; ++n)
#pragma unroll
            for (int i = 0; i < 8; ++i) acc[m][n][i] = 0.0f;

    // per-thread staging offsets
    const long aOff0 = (mBlock + ldRow) * (long)K_AL + ldCol;
    const long aOff1 = aOff0 + 64 * (long)K_AL;
    const long bOff0 = (long)(nBlock + ldRow) * K_AL + ldCol;
    const long bOff1 = bOff0 + 64 * (long)K_AL;
    const int  s0 = ldRow * LDSS + ldCol;
    const int  s1 = s0 + 64 * LDSS;

    // ---- prologue: stage tile 0 into buffer 0 ----
    bf16x8 rAh0 = *(const bf16x8*)&Ahi[aOff0];
    bf16x8 rAh1 = *(const bf16x8*)&Ahi[aOff1];
    bf16x8 rAl0 = *(const bf16x8*)&Alo[aOff0];
    bf16x8 rAl1 = *(const bf16x8*)&Alo[aOff1];
    bf16x8 rBh0 = *(const bf16x8*)&Bhi[bOff0];
    bf16x8 rBh1 = *(const bf16x8*)&Bhi[bOff1];
    bf16x8 rBl0 = *(const bf16x8*)&Blo[bOff0];
    bf16x8 rBl1 = *(const bf16x8*)&Blo[bOff1];
    *(bf16x8*)&sAh[0][s0] = rAh0;
    *(bf16x8*)&sAh[0][s1] = rAh1;
    *(bf16x8*)&sAl[0][s0] = rAl0;
    *(bf16x8*)&sAl[0][s1] = rAl1;
    *(bf16x8*)&sBh[0][s0] = rBh0;
    *(bf16x8*)&sBh[0][s1] = rBh1;
    *(bf16x8*)&sBl[0][s0] = rBl0;
    *(bf16x8*)&sBl[0][s1] = rBl1;

    int cur = 0;
    for (int kbi = 0; kbi < NKSTEPS; ++kbi) {
        __syncthreads();   // tile kbi visible in buffer `cur`; prev reads done

        // issue next tile's global loads first: latency hidden behind 24 WMMAs
        const int havenext = (kbi + 1 < NKSTEPS);
        if (havenext) {
            const long kOff = (long)(kbi + 1) * BK;
            rAh0 = *(const bf16x8*)&Ahi[aOff0 + kOff];
            rAh1 = *(const bf16x8*)&Ahi[aOff1 + kOff];
            rAl0 = *(const bf16x8*)&Alo[aOff0 + kOff];
            rAl1 = *(const bf16x8*)&Alo[aOff1 + kOff];
            rBh0 = *(const bf16x8*)&Bhi[bOff0 + kOff];
            rBh1 = *(const bf16x8*)&Bhi[bOff1 + kOff];
            rBl0 = *(const bf16x8*)&Blo[bOff0 + kOff];
            rBl1 = *(const bf16x8*)&Blo[bOff1 + kOff];
        }

        // ---- compute tile kbi from buffer `cur` ----
        Frag a_hi[2], a_lo[2];
#pragma unroll
        for (int m = 0; m < 2; ++m) {
            const int aBase = (wm * 32 + m * 16 + lr) * LDSS;
            a_hi[m].h[0] = *(const bf16x8*)&sAh[cur][aBase + ka0];
            a_hi[m].h[1] = *(const bf16x8*)&sAh[cur][aBase + ka0 + 16];
            a_lo[m].h[0] = *(const bf16x8*)&sAl[cur][aBase + ka0];
            a_lo[m].h[1] = *(const bf16x8*)&sAl[cur][aBase + ka0 + 16];
        }
#pragma unroll
        for (int n = 0; n < 4; ++n) {
            const int bBase = (wn * 64 + n * 16 + lr) * LDSS;
            Frag b_hi, b_lo;
            b_hi.h[0] = *(const bf16x8*)&sBh[cur][bBase + kb0];
            b_hi.h[1] = *(const bf16x8*)&sBh[cur][bBase + kb0 + 16];
            b_lo.h[0] = *(const bf16x8*)&sBl[cur][bBase + kb0];
            b_lo.h[1] = *(const bf16x8*)&sBl[cur][bBase + kb0 + 16];
#pragma unroll
            for (int m = 0; m < 2; ++m) {
                // fp32-emulating bf16x3: hi*hi + hi*lo + lo*hi, f32 accumulate
                acc[m][n] = __builtin_amdgcn_wmma_f32_16x16x32_bf16(
                    false, a_hi[m].v, false, b_hi.v, (short)0, acc[m][n], false, false);
                acc[m][n] = __builtin_amdgcn_wmma_f32_16x16x32_bf16(
                    false, a_hi[m].v, false, b_lo.v, (short)0, acc[m][n], false, false);
                acc[m][n] = __builtin_amdgcn_wmma_f32_16x16x32_bf16(
                    false, a_lo[m].v, false, b_hi.v, (short)0, acc[m][n], false, false);
            }
        }

        // ---- stage next tile into the other buffer ----
        if (havenext) {
            const int nxt = cur ^ 1;
            *(bf16x8*)&sAh[nxt][s0] = rAh0;
            *(bf16x8*)&sAh[nxt][s1] = rAh1;
            *(bf16x8*)&sAl[nxt][s0] = rAl0;
            *(bf16x8*)&sAl[nxt][s1] = rAl1;
            *(bf16x8*)&sBh[nxt][s0] = rBh0;
            *(bf16x8*)&sBh[nxt][s1] = rBh1;
            *(bf16x8*)&sBl[nxt][s0] = rBl0;
            *(bf16x8*)&sBl[nxt][s1] = rBl1;
        }
        cur ^= 1;
    }

    // epilogue: C layout -> VGPR i holds (M = i + 8*half, N = lr); add bias, scatter
    const long rowBase = mBlock + wm * 32 + half * 8;
#pragma unroll
    for (int m = 0; m < 2; ++m) {
#pragma unroll
        for (int n = 0; n < 4; ++n) {
            const int cn = nBlock + wn * 64 + n * 16 + lr;  // compact channel
            const int oc = alive_out[cn];                   // full-width channel
            const float bv = bias[oc];
#pragma unroll
            for (int i = 0; i < 8; ++i)
                out[(rowBase + m * 16 + i) * (long)D_MODEL + oc] = acc[m][n][i] + bv;
        }
    }
}

// ---------------- launch ----------------
extern "C" void kernel_launch(void* const* d_in, const int* in_sizes, int n_in,
                              void* d_out, int out_size, void* d_ws, size_t ws_size,
                              hipStream_t stream) {
    const float* x         = (const float*)d_in[0];
    const float* W         = (const float*)d_in[1];
    const float* b         = (const float*)d_in[2];
    const int*   alive_in  = (const int*)d_in[3];
    const int*   alive_out = (const int*)d_in[4];
    float*       out       = (float*)d_out;

    // workspace layout (all offsets 16B aligned)
    bf16* Ahi = (bf16*)d_ws;
    bf16* Alo = Ahi + (size_t)M_ROWS * K_AL;
    bf16* Bhi = Alo + (size_t)M_ROWS * K_AL;
    bf16* Blo = Bhi + (size_t)N_AL * K_AL;

    // 1) zero the full-width output (scatter target)
    {
        long n4 = (long)M_ROWS * D_MODEL / 4;
        int blocks = (int)((n4 + 255) / 256);
        zero_kernel<<<blocks, 256, 0, stream>>>((f32x4*)out, n4);
    }
    // 2) gather + split to bf16 hi/lo
    {
        long tot = (long)M_ROWS * K_AL;
        int blocks = (int)((tot + 255) / 256);
        gather_x_kernel<<<blocks, 256, 0, stream>>>(x, alive_in, Ahi, Alo);
    }
    {
        long tot = (long)N_AL * K_AL;
        int blocks = (int)((tot + 255) / 256);
        gather_w_kernel<<<blocks, 256, 0, stream>>>(W, alive_in, alive_out, Bhi, Blo);
    }
    // 3) WMMA GEMM + bias + scatter
    {
        dim3 grid(M_ROWS / BM, N_AL / BN);  // 64 x 16
        gemm_kernel<<<grid, 256, 0, stream>>>(Ahi, Alo, Bhi, Blo, b, alive_out, out);
    }
}
